// LTAE_9431748182688
// MI455X (gfx1250) — matrix operand order repaired and checked by reading, not compiled, not run on hardware
//
#include <hip/hip_runtime.h>

// LTAE fused kernel for gfx1250 (MI455X): one workgroup (8 wave32) per batch
// element. Whole activation tile lives in LDS (f16), GEMMs on
// v_wmma_f32_16x16x32_f16, raw input streamed with global_load_async_to_lds_b128
// (ASYNCcnt) double-buffered against compute.

typedef __attribute__((ext_vector_type(16))) _Float16 v16h;
typedef __attribute__((ext_vector_type(8)))  float    v8f;

#define SS    256   // seq len
#define CC    128   // in channels
#define DD    256   // d_model
#define HH    16    // heads
#define DKK   8     // d_k
#define DVV   16    // d_v
#define HSTR2 264   // hbuf row stride in halves (pad 8)
#define EPSF  1e-5f
#define XCH   64    // rows per x chunk

__device__ __forceinline__ float wave_sum(float v) {
#pragma unroll
  for (int o = 16; o > 0; o >>= 1) v += __shfl_xor(v, o, 32);
  return v;
}
__device__ __forceinline__ float wave_max(float v) {
#pragma unroll
  for (int o = 16; o > 0; o >>= 1) v = fmaxf(v, __shfl_xor(v, o, 32));
  return v;
}

// all 256 threads must call; reduces (a,b) over the block
__device__ __forceinline__ void block_reduce2(float& a, float& b, float* red, int tid) {
  a = wave_sum(a);
  b = wave_sum(b);
  const int w = tid >> 5, l = tid & 31;
  if (l == 0) { red[w] = a; red[8 + w] = b; }
  __syncthreads();
  if (tid == 0) {
    float sa = 0.f, sb = 0.f;
#pragma unroll
    for (int i = 0; i < 8; ++i) { sa += red[i]; sb += red[8 + i]; }
    red[16] = sa; red[17] = sb;
  }
  __syncthreads();
  a = red[16];
  b = red[17];
  __syncthreads();
}

// ---- async global -> LDS staging (CDNA5 ASYNCcnt path) ----
// copies XCH*CC f32 (32 KB) : 8 x b128 per thread, 256 threads
__device__ __forceinline__ void async_load_chunk(const float* gbase, float* lds_dst, int tid) {
  const unsigned long long sbase = (unsigned long long)(uintptr_t)gbase;
  unsigned int ldsoff = (unsigned int)(uintptr_t)lds_dst + tid * 16u;
  unsigned int voff   = tid * 16u;
#pragma unroll
  for (int j = 0; j < 8; ++j) {
    asm volatile("global_load_async_to_lds_b128 %0, %1, %2 offset:0"
                 :: "v"(ldsoff), "v"(voff), "s"(sbase) : "memory");
    ldsoff += 4096u;
    voff   += 4096u;
  }
}
__device__ __forceinline__ void wait_async0() {
  asm volatile("s_wait_asynccnt 0x0" ::: "memory");
}

// ---- WMMA fragment loaders (layouts per CDNA5 ISA 7.12.2, wave32) ----
// A (16x32 f16, row-major, leading dim ldk halves):
//   lane<16: M=lane, K in {0..7}u{16..23}; lane>=16: M=lane-16, K in {8..15}u{24..31}
__device__ __forceinline__ v16h load_a_f16(const _Float16* base, int ldk, int lane) {
  const int row = lane & 15;
  const int khi = (lane >> 4) * 8;
  const _Float16* p = base + row * ldk + khi;
  v16h a;
#pragma unroll
  for (int v = 0; v < 8; ++v) {
    const int k = 2 * v + (v >= 4 ? 8 : 0);
    a[2 * v]     = p[k];
    a[2 * v + 1] = p[k + 1];
  }
  return a;
}
// B (32x16 f16) from row-major [N][K] f16, leading dim ldk:
//   lane<16: N=lane, K in {0..7}u{16..23}; lane>=16: N=lane-16, K in {8..15}u{24..31}
__device__ __forceinline__ v16h load_b_f16(const _Float16* base, int ldk, int lane) {
  const int col = lane & 15;
  const int kb  = (lane >> 4) * 8;
  const _Float16* p = base + col * ldk + kb;
  v16h b;
#pragma unroll
  for (int i = 0; i < 16; ++i) b[i] = p[i + (i >= 8 ? 8 : 0)];
  return b;
}
__device__ __forceinline__ v8f wmma_f16(v16h a, v16h b, v8f c) {
  return __builtin_amdgcn_wmma_f32_16x16x32_f16(false, a, false, b, (short)0, c, false, false);
}

__global__ __launch_bounds__(256, 1) void ltae_fused_kernel(
    const float* __restrict__ x,        // (B,S,C)
    const float* __restrict__ ln_in_g, const float* __restrict__ ln_in_b,     // (C)
    const float* __restrict__ conv_w,  const float* __restrict__ conv_b,      // (D,C),(D)
    const float* __restrict__ ln_conv_g, const float* __restrict__ ln_conv_b, // (D,S)
    const float* __restrict__ Qw,                                             // (H,DK)
    const float* __restrict__ k_w, const float* __restrict__ k_b,             // (H*DK,D),(H*DK)
    const float* __restrict__ mlp_w, const float* __restrict__ mlp_b,         // (128,D),(128)
    const float* __restrict__ bn_g, const float* __restrict__ bn_b,
    const float* __restrict__ bn_mean, const float* __restrict__ bn_var,      // (128)
    const float* __restrict__ ln_out_g, const float* __restrict__ ln_out_b,   // (128)
    float* __restrict__ out)                                                  // (B,128)
{
  extern __shared__ char smem_raw[];
  char* sp = smem_raw;
  _Float16* hbuf  = (_Float16*)sp;  sp += SS * HSTR2 * 2;       // 135168 B
  _Float16* cw16  = (_Float16*)sp;  sp += DD * CC * 2;          //  65536 B
  float*    xst   = (float*)sp;     sp += 2 * XCH * CC * 4;     //  65536 B (double buffer)
  _Float16* abuf  = (_Float16*)sp;  sp += XCH * CC * 2;         //  16384 B
  float*    attn  = (float*)sp;     sp += SS * HH * 4;          //  16384 B
  _Float16* qkw16 = (_Float16*)sp;  sp += HH * DD * 2;          //   8192 B
  float*    outv  = (float*)sp;     sp += DD * 4;               //   1024 B
  float*    red   = (float*)sp;     sp += 32 * 4;               //    128 B
  float*    qkb   = (float*)sp;     sp += HH * 4;               //     64 B

  const int tid  = threadIdx.x;
  const int lane = tid & 31;
  const int wave = __builtin_amdgcn_readfirstlane(tid >> 5);  // provably uniform
  const float* xb = x + (size_t)blockIdx.x * SS * CC;

  // kick off async DMA of x chunk 0 while we convert weights
  async_load_chunk(xb, xst, tid);

  // ---------- stage 0: conv_w f32 -> f16 in LDS (done once per WG, reused 4x16x4 times) ----------
  {
    const float4* wp = (const float4*)(conv_w + (size_t)tid * CC);
    _Float16* cp = cw16 + tid * CC;
#pragma unroll 8
    for (int i = 0; i < CC / 4; ++i) {
      const float4 w4 = wp[i];
      cp[4 * i]     = (_Float16)w4.x;
      cp[4 * i + 1] = (_Float16)w4.y;
      cp[4 * i + 2] = (_Float16)w4.z;
      cp[4 * i + 3] = (_Float16)w4.w;
    }
  }

  // ---------- stage 1+2: input LN (over C) -> f16, conv GEMM via WMMA, chunked ----------
  for (int c = 0; c < SS / XCH; ++c) {
    float* xcur = xst + (c & 1) * (XCH * CC);
    wait_async0();        // chunk c resident in LDS
    __syncthreads();      // visible to all waves; prev reads of other buffer done
    if (c < SS / XCH - 1)
      async_load_chunk(xb + (size_t)(c + 1) * XCH * CC, xst + ((c + 1) & 1) * (XCH * CC), tid);

    // LN: one wave per row, lane handles 4 channels
    for (int r = wave; r < XCH; r += 8) {
      const float4 v = *(const float4*)(xcur + r * CC + lane * 4);
      float sum = v.x + v.y + v.z + v.w;
      float ssq = v.x * v.x + v.y * v.y + v.z * v.z + v.w * v.w;
      sum = wave_sum(sum);
      ssq = wave_sum(ssq);
      const float mu   = sum * (1.f / (float)CC);
      const float rstd = rsqrtf(ssq * (1.f / (float)CC) - mu * mu + EPSF);
      const float4 g  = *(const float4*)(ln_in_g + lane * 4);
      const float4 bb = *(const float4*)(ln_in_b + lane * 4);
      _Float16* ap = abuf + r * CC + lane * 4;
      ap[0] = (_Float16)((v.x - mu) * rstd * g.x + bb.x);
      ap[1] = (_Float16)((v.y - mu) * rstd * g.y + bb.y);
      ap[2] = (_Float16)((v.z - mu) * rstd * g.z + bb.z);
      ap[3] = (_Float16)((v.w - mu) * rstd * g.w + bb.w);
    }
    __syncthreads();

    // GEMM: 4 M-tiles x 16 N-tiles, K=C=128 (4 WMMA per tile); 8 jobs per wave
    for (int j = wave; j < 64; j += 8) {
      const int mt = j >> 4, nt = j & 15;
      v8f acc = {};
#pragma unroll
      for (int kb = 0; kb < 4; ++kb) {
        const v16h a  = load_a_f16(abuf + (mt * 16) * CC + kb * 32, CC, lane);
        const v16h bm = load_b_f16(cw16 + (nt * 16) * CC + kb * 32, CC, lane);
        acc = wmma_f16(a, bm, acc);
      }
      const int   n    = nt * 16 + (lane & 15);
      const float bias = conv_b[n];
      const int   mrow = c * XCH + mt * 16 + 8 * (lane >> 4);
#pragma unroll
      for (int r = 0; r < 8; ++r)
        hbuf[(mrow + r) * HSTR2 + n] = (_Float16)(acc[r] + bias);
    }
    __syncthreads();
  }

  // ---------- stage 3: LayerNorm over (S,D) jointly + positional encoding ----------
  {
    float sum = 0.f, ssq = 0.f;
    const _Float16* hr = hbuf + tid * HSTR2;  // thread t owns row s = t
#pragma unroll 4
    for (int d = 0; d < DD; ++d) { const float v = (float)hr[d]; sum += v; ssq += v * v; }
    block_reduce2(sum, ssq, red, tid);
    const float mu   = sum * (1.f / (float)(SS * DD));
    const float rstd = rsqrtf(ssq * (1.f / (float)(SS * DD)) - mu * mu + EPSF);
    float invf[DVV];
#pragma unroll
    for (int j = 0; j < DVV; ++j)
      invf[j] = __powf(1000.f, -(float)(2 * (j / 2)) / (float)DVV);
    const int   s  = tid;
    const float fp = (float)(s + 1);
    _Float16* hw = hbuf + s * HSTR2;
    for (int d = 0; d < DD; ++d) {
      const int   j   = d & (DVV - 1);
      const float ang = fp * invf[j];
      const float pos = (j & 1) ? __cosf(ang) : __sinf(ang);
      hw[d] = (_Float16)(((float)hw[d] - mu) * rstd * ln_conv_g[d * SS + s] +
                         ln_conv_b[d * SS + s] + pos);
    }
  }
  __syncthreads();

  // ---------- stage 4: fold shared query into k_w (N: 128 -> 16), logits GEMM ----------
  {
    const float scale = 0.35355339059327373f;  // 1/sqrt(DK)
    const int h  = tid >> 4;
    const int d0 = (tid & 15) * 16;
#pragma unroll
    for (int dd = 0; dd < 16; ++dd) {
      const int d = d0 + dd;
      float acc = 0.f;
#pragma unroll
      for (int k = 0; k < DKK; ++k)
        acc += Qw[h * DKK + k] * k_w[(size_t)(h * DKK + k) * DD + d];
      qkw16[h * DD + d] = (_Float16)(acc * scale);
    }
    if (tid < HH) {
      float acc = 0.f;
#pragma unroll
      for (int k = 0; k < DKK; ++k) acc += Qw[tid * DKK + k] * k_b[tid * DKK + k];
      qkb[tid] = acc * scale;
    }
  }
  __syncthreads();

  // logits: M=S (16 tiles), N=H (1 tile), K=D (8 WMMA steps); 2 jobs per wave
  for (int mt = wave; mt < 16; mt += 8) {
    v8f acc = {};
#pragma unroll
    for (int kb = 0; kb < 8; ++kb) {
      const v16h a  = load_a_f16(hbuf + (mt * 16) * HSTR2 + kb * 32, HSTR2, lane);
      const v16h bm = load_b_f16(qkw16 + kb * 32, DD, lane);
      acc = wmma_f16(a, bm, acc);
    }
    const int   hcol = lane & 15;
    const float bias = qkb[hcol];
    const int   mrow = mt * 16 + 8 * (lane >> 4);
#pragma unroll
    for (int r = 0; r < 8; ++r)
      attn[(mrow + r) * HH + hcol] = acc[r] + bias;
  }
  __syncthreads();

  // ---------- stage 5: softmax over S per head (wave handles 2 heads, lane-parallel in s) ----------
  for (int hh = 0; hh < 2; ++hh) {
    const int h = wave + hh * 8;
    float ev[8];
    float m = -1e30f;
#pragma unroll
    for (int i = 0; i < 8; ++i) {
      ev[i] = attn[(lane + 32 * i) * HH + h];
      m = fmaxf(m, ev[i]);
    }
    m = wave_max(m);
    float sum = 0.f;
#pragma unroll
    for (int i = 0; i < 8; ++i) { ev[i] = __expf(ev[i] - m); sum += ev[i]; }
    sum = wave_sum(sum);
    const float inv = 1.f / sum;
#pragma unroll
    for (int i = 0; i < 8; ++i) attn[(lane + 32 * i) * HH + h] = ev[i] * inv;
  }
  __syncthreads();

  // ---------- stage 6: attention pooling  out[h,dv] = sum_s p[h,s] * h[s, h*16+dv] ----------
  {
    const int h = tid >> 4, dv = tid & 15;
    float acc = 0.f;
    for (int s = 0; s < SS; ++s)
      acc += attn[s * HH + h] * (float)hbuf[s * HSTR2 + h * DVV + dv];
    outv[tid] = acc;
  }
  __syncthreads();

  // ---------- stage 7: MLP -> BN(eval) -> ReLU -> LayerNorm -> out ----------
  {
    float z = 0.f;
    if (tid < 128) {
      const float4* wp = (const float4*)(mlp_w + (size_t)tid * DD);
      float acc = 0.f;
#pragma unroll 4
      for (int i = 0; i < DD / 4; ++i) {
        const float4 w4 = wp[i];
        acc += w4.x * outv[4 * i] + w4.y * outv[4 * i + 1] +
               w4.z * outv[4 * i + 2] + w4.w * outv[4 * i + 3];
      }
      acc += mlp_b[tid];
      acc = (acc - bn_mean[tid]) * rsqrtf(bn_var[tid] + EPSF) * bn_g[tid] + bn_b[tid];
      z = fmaxf(acc, 0.f);
    }
    float sum = z, ssq = z * z;
    block_reduce2(sum, ssq, red, tid);  // threads >=128 contribute 0
    const float mu   = sum * (1.f / 128.f);
    const float rstd = rsqrtf(ssq * (1.f / 128.f) - mu * mu + EPSF);
    if (tid < 128)
      out[(size_t)blockIdx.x * 128 + tid] = (z - mu) * rstd * ln_out_g[tid] + ln_out_b[tid];
  }
}

extern "C" void kernel_launch(void* const* d_in, const int* in_sizes, int n_in,
                              void* d_out, int out_size, void* d_ws, size_t ws_size,
                              hipStream_t stream) {
  (void)n_in; (void)out_size; (void)d_ws; (void)ws_size;
  const float* x         = (const float*)d_in[0];
  const float* ln_in_g   = (const float*)d_in[1];
  const float* ln_in_b   = (const float*)d_in[2];
  const float* conv_w    = (const float*)d_in[3];
  const float* conv_b    = (const float*)d_in[4];
  const float* ln_conv_g = (const float*)d_in[5];
  const float* ln_conv_b = (const float*)d_in[6];
  const float* Qw        = (const float*)d_in[7];
  const float* k_w       = (const float*)d_in[8];
  const float* k_b       = (const float*)d_in[9];
  const float* mlp_w     = (const float*)d_in[10];
  const float* mlp_b     = (const float*)d_in[11];
  const float* bn_g      = (const float*)d_in[12];
  const float* bn_b      = (const float*)d_in[13];
  const float* bn_mean   = (const float*)d_in[14];
  const float* bn_var    = (const float*)d_in[15];
  const float* ln_out_g  = (const float*)d_in[16];
  const float* ln_out_b  = (const float*)d_in[17];
  float* out = (float*)d_out;

  const int B = in_sizes[0] / (SS * CC);  // 1024

  const size_t smem = (size_t)SS * HSTR2 * 2   // hbuf f16
                    + (size_t)DD * CC * 2      // cw16
                    + (size_t)2 * XCH * CC * 4 // x stage (double buffer)
                    + (size_t)XCH * CC * 2     // abuf
                    + (size_t)SS * HH * 4      // attn
                    + (size_t)HH * DD * 2      // qkw16
                    + (size_t)DD * 4           // outv
                    + 32 * 4 + HH * 4;         // red + qkb  (~308.4 KB)
  hipFuncSetAttribute(reinterpret_cast<const void*>(ltae_fused_kernel),
                      hipFuncAttributeMaxDynamicSharedMemorySize, (int)smem);

  ltae_fused_kernel<<<B, 256, smem, stream>>>(
      x, ln_in_g, ln_in_b, conv_w, conv_b, ln_conv_g, ln_conv_b,
      Qw, k_w, k_b, mlp_w, mlp_b, bn_g, bn_b, bn_mean, bn_var,
      ln_out_g, ln_out_b, out);
}
